// CrossAttention_17987323035751
// MI455X (gfx1250) — matrix operand order repaired and checked by reading, not compiled, not run on hardware
//
#include <hip/hip_runtime.h>
#include <hip/hip_bf16.h>

#define DIMC   1024
#define HEADS  16
#define HDIM   64
#define BATCH  4
#define NQ     4096
#define TCTX   256
#define TP     288      // kv padded to 9 chunks of 32
#define TVALID 257      // T + null token
#define SCALE_F 8.0f
#define LN_EPS_F 1e-5f
#define NORM_EPS_F 1e-12f

typedef __attribute__((ext_vector_type(8)))  _Float16 v8h;
typedef __attribute__((ext_vector_type(16))) _Float16 v16h;
typedef __attribute__((ext_vector_type(8)))  float    v8f;

__device__ __forceinline__ v16h cat8(v8h a, v8h b) {
  return __builtin_shufflevector(a, b, 0,1,2,3,4,5,6,7,8,9,10,11,12,13,14,15);
}
__device__ __forceinline__ v8f wmma16(v16h a, v16h b, v8f c) {
  return __builtin_amdgcn_wmma_f32_16x16x32_f16(false, a, false, b, (short)0, c, false, false);
}
__device__ __forceinline__ float redmax16(float t) {
  t = fmaxf(t, __shfl_xor(t, 1, 32));
  t = fmaxf(t, __shfl_xor(t, 2, 32));
  t = fmaxf(t, __shfl_xor(t, 4, 32));
  t = fmaxf(t, __shfl_xor(t, 8, 32));
  return t;
}
__device__ __forceinline__ float redsum16(float t) {
  t += __shfl_xor(t, 1, 32);
  t += __shfl_xor(t, 2, 32);
  t += __shfl_xor(t, 4, 32);
  t += __shfl_xor(t, 8, 32);
  return t;
}
__device__ __forceinline__ float redsum32(float t) {
  t += __shfl_xor(t, 1, 32);
  t += __shfl_xor(t, 2, 32);
  t += __shfl_xor(t, 4, 32);
  t += __shfl_xor(t, 8, 32);
  t += __shfl_xor(t, 16, 32);
  return t;
}

// ---------------- LayerNorm (row length 1024), optional f16 output ----------
template <typename OutT>
__global__ void __launch_bounds__(256)
ln_kernel(const float* __restrict__ X, const float* __restrict__ g,
          const float* __restrict__ bb, OutT* __restrict__ Y) {
  const int row = blockIdx.x;
  const float* xr = X + (size_t)row * DIMC;
  __shared__ float rs[8], rs2[8];
  float vals[4];
  float s = 0.f, s2 = 0.f;
#pragma unroll
  for (int i = 0; i < 4; ++i) {
    float v = xr[threadIdx.x + i * 256];
    vals[i] = v; s += v; s2 += v * v;
  }
  s = redsum32(s); s2 = redsum32(s2);
  const int lane = threadIdx.x & 31, wave = threadIdx.x >> 5;
  if (lane == 0) { rs[wave] = s; rs2[wave] = s2; }
  __syncthreads();
  float tot = 0.f, tot2 = 0.f;
#pragma unroll
  for (int i = 0; i < 8; ++i) { tot += rs[i]; tot2 += rs2[i]; }
  const float mean = tot * (1.0f / DIMC);
  const float var  = tot2 * (1.0f / DIMC) - mean * mean;
  const float rstd = rsqrtf(var + LN_EPS_F);
  OutT* yr = Y + (size_t)row * DIMC;
#pragma unroll
  for (int i = 0; i < 4; ++i) {
    int c = threadIdx.x + i * 256;
    yr[c] = (OutT)((vals[i] - mean) * rstd * g[c] + bb[c]);
  }
}

// ---------------- Weight transpose + cast: Wt[n][k] = (f16) W[k][n] ---------
__global__ void __launch_bounds__(256)
twcast_kernel(const float* __restrict__ W, _Float16* __restrict__ Wt,
              int K, int N) {
  size_t idx = (size_t)blockIdx.x * 256 + threadIdx.x;
  if (idx >= (size_t)K * N) return;
  int k = (int)(idx / N), n = (int)(idx % N);
  Wt[(size_t)n * K + k] = (_Float16)W[idx];
}

// ---------------- WMMA GEMM: C[M,N] = A[M,K](f16) * Bt[N,K](f16) ------------
// Each wave: one 16(M) x 64(N) strip, 4 f32 accumulators. K multiple of 32.
__global__ void __launch_bounds__(256)
wmma_gemm_kernel(const _Float16* __restrict__ A, const _Float16* __restrict__ Bt,
                 float* __restrict__ C, int M, int N, int K) {
  const int lane = threadIdx.x & 31, wave = threadIdx.x >> 5;
  const int tiles_n = N >> 6;
  const int tile = blockIdx.x * 8 + wave;
  const int tm = tile / tiles_n;
  const int tn = tile % tiles_n;
  if (tm >= (M >> 4)) return;
  const int col   = lane & 15;
  const int koff  = (lane >> 4) << 4;   // A: 0 or 16
  const int kboff = (lane >> 4) << 4;   // B: 0 or 16
  const _Float16* arow = A + (size_t)((tm << 4) + col) * K;
  v8f acc[4] = {};
  for (int k = 0; k < K; k += 32) {
    v8h a0 = *(const v8h*)(arow + k + koff);
    v8h a1 = *(const v8h*)(arow + k + koff + 8);
    v16h af = cat8(a0, a1);
#pragma unroll
    for (int j = 0; j < 4; ++j) {
      const int cn = (tn << 6) + (j << 4) + col;
      v16h bf = *(const v16h*)(Bt + (size_t)cn * K + k + kboff);
      acc[j] = wmma16(af, bf, acc[j]);
    }
  }
  const int mrow = (tm << 4) + ((lane >> 4) << 3);
#pragma unroll
  for (int j = 0; j < 4; ++j) {
    float* cp = C + (size_t)mrow * N + (tn << 6) + (j << 4) + col;
#pragma unroll
    for (int i = 0; i < 8; ++i) cp[(size_t)i * N] = acc[j][i];
  }
}

// ---------------- Q prep: l2-normalize per head, fold q_scale * SCALE -------
// Q f32 [B,N,INNER] -> Qh f16 [B,H,N,64]; one wave per (b,h,n) row.
__global__ void __launch_bounds__(256)
qprep_kernel(const float* __restrict__ Q, const float* __restrict__ qs,
             _Float16* __restrict__ Qh) {
  const int lane = threadIdx.x & 31, wave = threadIdx.x >> 5;
  const long r = (long)blockIdx.x * 8 + wave;     // over B*H*N
  if (r >= (long)BATCH * HEADS * NQ) return;
  const int n = (int)(r % NQ);
  const int h = (int)((r / NQ) % HEADS);
  const int b = (int)(r / ((long)NQ * HEADS));
  const float* src = Q + ((size_t)b * NQ + n) * DIMC + h * HDIM;
  const int d0 = lane * 2;
  float f0 = src[d0], f1 = src[d0 + 1];
  float ss = redsum32(f0 * f0 + f1 * f1);
  float inv = SCALE_F / fmaxf(sqrtf(ss), NORM_EPS_F);
  _Float16* dst = Qh + (((size_t)b * HEADS + h) * NQ + n) * HDIM;
  dst[d0]     = (_Float16)(f0 * inv * qs[d0]);
  dst[d0 + 1] = (_Float16)(f1 * inv * qs[d0 + 1]);
}

// ---------------- KV prep: null token, pad to TP, normalize k ---------------
// KV f32 [B,T,2*INNER] -> Kh f16 [B,H,TP,64] (row major), Vt f16 [B,H,64,TP].
__global__ void __launch_bounds__(256)
kvprep_kernel(const float* __restrict__ KV, const float* __restrict__ nullkv,
              const float* __restrict__ ks, _Float16* __restrict__ Kh,
              _Float16* __restrict__ Vt) {
  const int lane = threadIdx.x & 31, wave = threadIdx.x >> 5;
  const long r = (long)blockIdx.x * 8 + wave;     // over B*H*TP
  if (r >= (long)BATCH * HEADS * TP) return;
  const int t = (int)(r % TP);
  const int h = (int)((r / TP) % HEADS);
  const int b = (int)(r / ((long)TP * HEADS));
  const int d0 = lane * 2;
  float k0 = 0.f, k1 = 0.f, vv0 = 0.f, vv1 = 0.f;
  if (t < TCTX) {
    const float* src = KV + ((size_t)b * TCTX + t) * (2 * DIMC) + h * HDIM;
    k0 = src[d0]; k1 = src[d0 + 1];
    vv0 = src[DIMC + d0]; vv1 = src[DIMC + d0 + 1];
  } else if (t == TCTX) {
    k0 = nullkv[d0]; k1 = nullkv[d0 + 1];
    vv0 = nullkv[HDIM + d0]; vv1 = nullkv[HDIM + d0 + 1];
  }
  float ss = redsum32(k0 * k0 + k1 * k1);
  float inv = 1.0f / fmaxf(sqrtf(ss), NORM_EPS_F);
  const size_t bh = (size_t)b * HEADS + h;
  _Float16* kd = Kh + (bh * TP + t) * HDIM;
  kd[d0]     = (_Float16)(k0 * inv * ks[d0]);
  kd[d0 + 1] = (_Float16)(k1 * inv * ks[d0 + 1]);
  _Float16* vd = Vt + bh * (size_t)HDIM * TP + t;   // d-major
  vd[(size_t)d0 * TP]       = (_Float16)vv0;
  vd[(size_t)(d0 + 1) * TP] = (_Float16)vv1;
}

// ---------------- Flash attention: 1 wave = 16 q rows, 9 kv chunks of 32 ----
__global__ void __launch_bounds__(256)
attn_kernel(const _Float16* __restrict__ Qh, const _Float16* __restrict__ Kh,
            const _Float16* __restrict__ Vt, _Float16* __restrict__ Out) {
  __shared__ _Float16 lP[8][16][34];
  const int lane = threadIdx.x & 31, wave = threadIdx.x >> 5;
  const int qblocks = NQ / 128;
  const int bh = blockIdx.x / qblocks;
  const int qb = blockIdx.x % qblocks;
  const int b = bh / HEADS, h = bh % HEADS;
  const int qrow = qb * 128 + wave * 16;
  const int col   = lane & 15;
  const int half  = lane >> 4;
  const int koff  = half << 4;
  const int kboff = half << 4;

  // A-fragments of q (d 0..31 and d 32..63)
  const _Float16* qr = Qh + (((size_t)bh * NQ) + qrow + col) * HDIM;
  v16h qa0 = cat8(*(const v8h*)(qr + koff),      *(const v8h*)(qr + koff + 8));
  v16h qa1 = cat8(*(const v8h*)(qr + 32 + koff), *(const v8h*)(qr + 32 + koff + 8));

  const _Float16* kbh = Kh + (size_t)bh * TP * HDIM;
  const _Float16* vbh = Vt + (size_t)bh * HDIM * TP;

  float mrun[8], lrun[8];
#pragma unroll
  for (int i = 0; i < 8; ++i) { mrun[i] = -3.0e38f; lrun[i] = 0.f; }
  v8f acc[4] = {};

  for (int c = 0; c < TP / 32; ++c) {
    const int kv0 = c * 32;
    // ---- scores: two 16x16 column tiles (kv0..+15, kv0+16..+31)
    v8f s0 = {}, s1 = {};
    {
      const _Float16* kp = kbh + (size_t)(kv0 + col) * HDIM;
      s0 = wmma16(qa0, *(const v16h*)(kp + kboff), s0);
      s0 = wmma16(qa1, *(const v16h*)(kp + 32 + kboff), s0);
      const _Float16* kp2 = kbh + (size_t)(kv0 + 16 + col) * HDIM;
      s1 = wmma16(qa0, *(const v16h*)(kp2 + kboff), s1);
      s1 = wmma16(qa1, *(const v16h*)(kp2 + 32 + kboff), s1);
    }
    // ---- mask padded kv positions
    if (kv0 + col >= TVALID) {
#pragma unroll
      for (int i = 0; i < 8; ++i) s0[i] = -3.0e38f;
    }
    if (kv0 + 16 + col >= TVALID) {
#pragma unroll
      for (int i = 0; i < 8; ++i) s1[i] = -3.0e38f;
    }
    // ---- online softmax (row stats live per half-wave, per C-register)
    float p0[8], p1[8];
#pragma unroll
    for (int i = 0; i < 8; ++i) {
      float t = redmax16(fmaxf(s0[i], s1[i]));
      float mnew = fmaxf(mrun[i], t);
      float sc = __expf(mrun[i] - mnew);
      p0[i] = __expf(s0[i] - mnew);
      p1[i] = __expf(s1[i] - mnew);
      float rsum = redsum16(p0[i] + p1[i]);
      lrun[i] = lrun[i] * sc + rsum;
      mrun[i] = mnew;
#pragma unroll
      for (int j = 0; j < 4; ++j) acc[j][i] = acc[j][i] * sc;
    }
    // ---- transpose P (C-layout) -> A-layout via LDS
#pragma unroll
    for (int i = 0; i < 8; ++i) {
      const int m = half * 8 + i;
      lP[wave][m][col]      = (_Float16)p0[i];
      lP[wave][m][col + 16] = (_Float16)p1[i];
    }
    asm volatile("s_wait_dscnt 0" ::: "memory");
    v16h pa;
#pragma unroll
    for (int i = 0; i < 8; ++i) {
      pa[i]     = lP[wave][col][koff + i];
      pa[8 + i] = lP[wave][col][koff + 8 + i];
    }
    // ---- O += P(16x32) @ V(32x64): V d-major so B-fragments are contiguous
#pragma unroll
    for (int j = 0; j < 4; ++j) {
      const _Float16* vp = vbh + (size_t)(j * 16 + col) * TP + kv0 + kboff;
      acc[j] = wmma16(pa, *(const v16h*)vp, acc[j]);
    }
  }
  // ---- normalize and write f16 [B,N,H*64] for the Wo GEMM
  float invl[8];
#pragma unroll
  for (int i = 0; i < 8; ++i) invl[i] = 1.0f / lrun[i];
#pragma unroll
  for (int j = 0; j < 4; ++j) {
    const int dcol = h * HDIM + j * 16 + col;
#pragma unroll
    for (int i = 0; i < 8; ++i) {
      const int row = qrow + half * 8 + i;
      Out[((size_t)b * NQ + row) * DIMC + dcol] = (_Float16)(acc[j][i] * invl[i]);
    }
  }
}

// ============================== launch ======================================
extern "C" void kernel_launch(void* const* d_in, const int* in_sizes, int n_in,
                              void* d_out, int out_size, void* d_ws, size_t ws_size,
                              hipStream_t stream) {
  const float* x       = (const float*)d_in[0];
  const float* context = (const float*)d_in[1];
  const float* Wq      = (const float*)d_in[2];
  const float* Wkv     = (const float*)d_in[3];
  const float* Wo      = (const float*)d_in[4];
  const float* null_kv = (const float*)d_in[5];
  const float* q_scale = (const float*)d_in[6];
  const float* k_scale = (const float*)d_in[7];
  const float* ln_in_g  = (const float*)d_in[8];
  const float* ln_in_b  = (const float*)d_in[9];
  const float* ln_ctx_g = (const float*)d_in[10];
  const float* ln_ctx_b = (const float*)d_in[11];
  const float* ln_out_g = (const float*)d_in[12];
  const float* ln_out_b = (const float*)d_in[13];
  float* out = (float*)d_out;

  char* ws = (char*)d_ws;
  size_t off = 0;
  auto take = [&](size_t bytes) -> char* {
    char* p = ws + off;
    off = (off + bytes + 255) & ~(size_t)255;
    return p;
  };
  const size_t MROWS = (size_t)BATCH * NQ;          // 16384
  _Float16* xn   = (_Float16*)take(MROWS * DIMC * 2);           // also attn-out (alias)
  _Float16* cn   = (_Float16*)take((size_t)BATCH * TCTX * DIMC * 2);
  _Float16* wqT  = (_Float16*)take((size_t)DIMC * DIMC * 2);
  _Float16* wkvT = (_Float16*)take((size_t)DIMC * 2 * DIMC * 2);
  _Float16* woT  = (_Float16*)take((size_t)DIMC * DIMC * 2);
  float*    qf   = (float*)take(MROWS * DIMC * 4);              // also final GEMM out (alias)
  float*    kvf  = (float*)take((size_t)BATCH * TCTX * 2 * DIMC * 4);
  _Float16* qh   = (_Float16*)take((size_t)BATCH * HEADS * NQ * HDIM * 2);
  _Float16* kh   = (_Float16*)take((size_t)BATCH * HEADS * TP * HDIM * 2);
  _Float16* vt   = (_Float16*)take((size_t)BATCH * HEADS * HDIM * TP * 2);
  _Float16* attn = xn;     // xn dead after Q GEMM
  float*    yf   = qf;     // qf dead after qprep

  // 1) LayerNorms -> f16
  ln_kernel<_Float16><<<(int)MROWS, 256, 0, stream>>>(x, ln_in_g, ln_in_b, xn);
  ln_kernel<_Float16><<<BATCH * TCTX, 256, 0, stream>>>(context, ln_ctx_g, ln_ctx_b, cn);

  // 2) Weight transpose + cast
  twcast_kernel<<<(DIMC * DIMC) / 256, 256, 0, stream>>>(Wq, wqT, DIMC, DIMC);
  twcast_kernel<<<(DIMC * 2 * DIMC) / 256, 256, 0, stream>>>(Wkv, wkvT, DIMC, 2 * DIMC);
  twcast_kernel<<<(DIMC * DIMC) / 256, 256, 0, stream>>>(Wo, woT, DIMC, DIMC);

  // 3) Projections (WMMA)
  wmma_gemm_kernel<<<(int)((MROWS / 16) * (DIMC / 64) / 8), 256, 0, stream>>>(
      xn, wqT, qf, (int)MROWS, DIMC, DIMC);
  wmma_gemm_kernel<<<(BATCH * TCTX / 16) * (2 * DIMC / 64) / 8, 256, 0, stream>>>(
      cn, wkvT, kvf, BATCH * TCTX, 2 * DIMC, DIMC);

  // 4) QK norm prep
  qprep_kernel<<<(BATCH * HEADS * NQ) / 8, 256, 0, stream>>>(qf, q_scale, qh);
  kvprep_kernel<<<(BATCH * HEADS * TP) / 8, 256, 0, stream>>>(kvf, null_kv, k_scale, kh, vt);

  // 5) Attention (WMMA flash)
  attn_kernel<<<BATCH * HEADS * (NQ / 128), 256, 0, stream>>>(qh, kh, vt, attn);

  // 6) Output projection (WMMA) + final LayerNorm
  wmma_gemm_kernel<<<(int)((MROWS / 16) * (DIMC / 64) / 8), 256, 0, stream>>>(
      attn, woT, yf, (int)MROWS, DIMC, DIMC);
  ln_kernel<float><<<(int)MROWS, 256, 0, stream>>>(yf, ln_out_g, ln_out_b, out);
}
